// GCNClusterNet_55353538511098
// MI455X (gfx1250) — compile-verified, matrix operand order checked by reading, not run on hardware
//
#include <hip/hip_runtime.h>
#include <hip/hip_bf16.h>
#include <math.h>

// ---------------------------------------------------------------------------
// Types for CDNA5 WMMA (wave32, 16x16x32 bf16 -> f32)
// ---------------------------------------------------------------------------
typedef __attribute__((ext_vector_type(16))) __bf16 v16bf;
typedef __attribute__((ext_vector_type(8)))  float  v8f;

#define NN      100000
#define NFEAT   128
#define NHID    512
#define NOUTP   64     // padded NOUT
#define NOUT    50
#define KCLUS   10
#define TEMP    100.0f

// ---------------------------------------------------------------------------
// Pack a row-major f32 weight matrix B [KDIM x nvalid] (row stride ldb) into
// fragment-ordered bf16: Bp[((ntile*KB + kb)*32 + lane)] is the v16bf fragment
// lane `lane` feeds to v_wmma_f32_16x16x32_bf16 for output-col tile `ntile`,
// K-step `kb`. Columns >= nvalid are zero-padded.
// One thread per fragment (= 16 bf16 = one aligned 32B store).
// ---------------------------------------------------------------------------
__global__ void pack_b_frag(const float* __restrict__ B, __bf16* __restrict__ Bp,
                            int KB, int ldb, int nvalid, int total)
{
    const int t = blockIdx.x * blockDim.x + threadIdx.x;
    if (t >= total) return;
    const int lane  = t & 31;
    const int kb    = (t >> 5) % KB;
    const int ntile = (t >> 5) / KB;
    const int col   = ntile * 16 + (lane & 15);
    const int khalf = (lane >> 4) * 8;
    const bool cv   = (col < nvalid);
    v16bf v;
#pragma unroll
    for (int j = 0; j < 8; ++j) {
        const int k0 = kb * 32 + khalf + j;
        const int k1 = k0 + 16;
        v[j]     = cv ? (__bf16)B[(size_t)k0 * ldb + col] : (__bf16)0.0f;
        v[j + 8] = cv ? (__bf16)B[(size_t)k1 * ldb + col] : (__bf16)0.0f;
    }
    ((v16bf*)Bp)[t] = v;
}

// ---------------------------------------------------------------------------
// Tiled GEMM via v_wmma_f32_16x16x32_bf16, pre-packed B.
// A:  [M x KDIM] f32 row-major (M = gridDim.x*16 exactly)
// Bp: fragment-ordered bf16 (see pack_b_frag)
// C:  [M x ldc] f32
// Each wave computes NTPW adjacent 16x16 tiles: one A fragment is reused for
// NTPW back-to-back WMMAs.
// ---------------------------------------------------------------------------
template<int KDIM, int NTPW, int WAVES>
__global__ __launch_bounds__(32 * WAVES)
void gemm_wmma_bf16(const float* __restrict__ A,
                    const __bf16* __restrict__ Bp,
                    float* __restrict__ C, int ldc)
{
    constexpr int KB = KDIM / 32;
    const int lane  = threadIdx.x & 31;
    const int wave  = threadIdx.x >> 5;
    const int mtile = blockIdx.x;
    const int nt0   = (blockIdx.y * WAVES + wave) * NTPW;

    const int rowA  = mtile * 16 + (lane & 15);
    const int khalf = (lane >> 4) * 8;

    const v16bf* bp = (const v16bf*)Bp;

    v8f acc[NTPW];
#pragma unroll
    for (int t = 0; t < NTPW; ++t) acc[t] = (v8f){};

    for (int kb = 0; kb < KB; ++kb) {
        v16bf a;
        const float* arow = A + (size_t)rowA * KDIM + kb * 32 + khalf;
#pragma unroll
        for (int j = 0; j < 8; ++j) {
            a[j]     = (__bf16)arow[j];           // K = kb*32+khalf+j
            a[j + 8] = (__bf16)arow[16 + j];      // K = kb*32+khalf+16+j
        }
#pragma unroll
        for (int t = 0; t < NTPW; ++t) {
            const v16bf b = bp[((size_t)(nt0 + t) * KB + kb) * 32 + lane];
            acc[t] = __builtin_amdgcn_wmma_f32_16x16x32_bf16(
                false, a, false, b, (short)0, acc[t], false, false);
        }
    }

    // C/D layout: lane<16 -> rows 0..7, lane>=16 -> rows 8..15 (col = lane&15)
    const int rbase = (lane >> 4) * 8;
#pragma unroll
    for (int t = 0; t < NTPW; ++t) {
        float* crow = C + (size_t)(mtile * 16 + rbase) * ldc + (nt0 + t) * 16 + (lane & 15);
#pragma unroll
        for (int r = 0; r < 8; ++r)
            crow[(size_t)r * ldc] = acc[t][r];
    }
}

// ---------------------------------------------------------------------------
// SpMM scatter: agg[rows[e], :] += vals[e] * sup[cols[e], :]
// One block per edge; native f32 global atomics.
// ---------------------------------------------------------------------------
__global__ void spmm_atomic(const int* __restrict__ erows,
                            const int* __restrict__ ecols,
                            const float* __restrict__ evals,
                            const float* __restrict__ sup,
                            float* __restrict__ agg,
                            int ld, int nf)
{
    const int e = blockIdx.x;
    const int r = erows[e];
    const int c = ecols[e];
    const float v = evals[e];
    const float* s = sup + (size_t)c * ld;
    float* d = agg + (size_t)r * ld;
    for (int f = threadIdx.x; f < nf; f += blockDim.x)
        atomicAdd(&d[f], v * s[f]);
}

// h = relu(agg + b1), in place. ld must be power of two (512).
__global__ void bias_relu_inplace(float* __restrict__ h,
                                  const float* __restrict__ b,
                                  long n, int ldmask)
{
    long i = (long)blockIdx.x * blockDim.x + threadIdx.x;
    if (i < n) {
        float t = h[i] + b[(int)(i & ldmask)];
        h[i] = t > 0.0f ? t : 0.0f;
    }
}

// embeds = agg2 + b2 -> d_out; data = row-normalized embeds (64-padded, pad=0)
__global__ void finalize_embeds(const float* __restrict__ agg2,
                                const float* __restrict__ b2,
                                float* __restrict__ emb_out,
                                float* __restrict__ data, int n)
{
    const int lane = threadIdx.x & 31;
    const int wave = threadIdx.x >> 5;
    const int i = blockIdx.x * (blockDim.x >> 5) + wave;
    if (i >= n) return;
    const float* arow = agg2 + (size_t)i * NOUTP;
    const int f0 = lane, f1 = lane + 32;
    float e0 = 0.0f, e1 = 0.0f;
    if (f0 < NOUT) e0 = arow[f0] + b2[f0];
    if (f1 < NOUT) e1 = arow[f1] + b2[f1];
    float ss = e0 * e0 + e1 * e1;
#pragma unroll
    for (int off = 16; off > 0; off >>= 1) ss += __shfl_xor(ss, off);
    const float inv = rsqrtf(ss);
    if (f0 < NOUT) emb_out[(size_t)i * NOUT + f0] = e0;
    if (f1 < NOUT) emb_out[(size_t)i * NOUT + f1] = e1;
    data[(size_t)i * NOUTP + f0] = e0 * inv;   // pad lanes write 0*inv = 0
    data[(size_t)i * NOUTP + f1] = e1 * inv;
}

// pad mu [10x50] -> [10x64] (pad cols zero)
__global__ void mu_pad(const float* __restrict__ mu, float* __restrict__ mup)
{
    int t = threadIdx.x;
    if (t < KCLUS * NOUTP) {
        int k = t / NOUTP, d = t % NOUTP;
        mup[t] = (d < NOUT) ? mu[k * NOUT + d] : 0.0f;
    }
}

__global__ void mu_unpad(const float* __restrict__ mup, float* __restrict__ out)
{
    int t = threadIdx.x;
    if (t < KCLUS * NOUT) {
        int k = t / NOUT, d = t % NOUT;
        out[t] = mup[k * NOUTP + d];
    }
}

// logits = data @ mu.T ; softmax(TEMP*logits) -> rout ; logits -> distout
__global__ __launch_bounds__(256)
void dist_softmax(const float* __restrict__ data,
                  const float* __restrict__ mu,        // [10*64] padded
                  float* __restrict__ rout,            // [n*10] or null
                  float* __restrict__ distout,         // [n*10] or null
                  int n)
{
    __shared__ float smu[KCLUS * NOUTP];
    for (int t = threadIdx.x; t < KCLUS * NOUTP; t += blockDim.x) smu[t] = mu[t];
    __syncthreads();
    const int i = blockIdx.x * blockDim.x + threadIdx.x;
    if (i >= n) return;
    const float* drow = data + (size_t)i * NOUTP;
    float logits[KCLUS];
    float mx = -3.0e38f;
#pragma unroll
    for (int k = 0; k < KCLUS; ++k) {
        float s = 0.0f;
#pragma unroll
        for (int f = 0; f < NOUTP; ++f) s += drow[f] * smu[k * NOUTP + f];
        logits[k] = s;
        mx = fmaxf(mx, TEMP * s);
    }
    float den = 0.0f, p[KCLUS];
#pragma unroll
    for (int k = 0; k < KCLUS; ++k) { p[k] = __expf(TEMP * logits[k] - mx); den += p[k]; }
    const float invd = 1.0f / den;
#pragma unroll
    for (int k = 0; k < KCLUS; ++k) {
        if (rout)    rout[(size_t)i * KCLUS + k] = p[k] * invd;
        if (distout) distout[(size_t)i * KCLUS + k] = logits[k];
    }
}

// acc[k,d] += sum_i r[i,k]*data[i,d]; rsum[k] += sum_i r[i,k]
__global__ __launch_bounds__(640)
void mu_reduce(const float* __restrict__ r,
               const float* __restrict__ data,
               float* __restrict__ acc,     // [640]
               float* __restrict__ rsum,    // [10]
               int n)
{
    const int k = threadIdx.x / NOUTP;
    const int d = threadIdx.x % NOUTP;
    const int rows_per = (n + gridDim.x - 1) / gridDim.x;
    const int i0 = blockIdx.x * rows_per;
    const int i1 = min(n, i0 + rows_per);
    float a = 0.0f, rs = 0.0f;
    for (int i = i0; i < i1; ++i) {
        const float rv = r[(size_t)i * KCLUS + k];
        a += rv * data[(size_t)i * NOUTP + d];
        if (d == 0) rs += rv;
    }
    atomicAdd(&acc[threadIdx.x], a);
    if (d == 0) atomicAdd(&rsum[k], rs);
}

__global__ void mu_finalize(const float* __restrict__ acc,
                            const float* __restrict__ rsum,
                            float* __restrict__ mu_out)
{
    int t = threadIdx.x;
    if (t < KCLUS * NOUTP) mu_out[t] = acc[t] / rsum[t / NOUTP];
}

// ---------------------------------------------------------------------------
extern "C" void kernel_launch(void* const* d_in, const int* in_sizes, int n_in,
                              void* d_out, int out_size, void* d_ws, size_t ws_size,
                              hipStream_t stream) {
    const float* x     = (const float*)d_in[0];
    const int*   erows = (const int*)  d_in[1];
    const int*   ecols = (const int*)  d_in[2];
    const float* evals = (const float*)d_in[3];
    const float* mu0   = (const float*)d_in[4];
    const float* W1    = (const float*)d_in[5];
    const float* b1    = (const float*)d_in[6];
    const float* W2    = (const float*)d_in[7];
    const float* b2    = (const float*)d_in[8];
    const int E = in_sizes[1];

    float* out     = (float*)d_out;
    float* out_mu  = out;                                   // [10*50]
    float* out_r   = out + KCLUS * NOUT;                    // [N*10]
    float* out_emb = out_r + (size_t)NN * KCLUS;            // [N*50]
    float* out_dst = out_emb + (size_t)NN * NOUT;           // [N*10]

    float* ws   = (float*)d_ws;
    float* sup1 = ws;                               // [N*512]  (dead after spmm1)
    float* agg1 = ws + (size_t)NN * NHID;           // [N*512]  = h after relu
    float* sup2 = ws;                               // [N*64]   reuses sup1 space
    float* agg2 = ws + (size_t)NN * NOUTP;          // [N*64]
    float* data = ws + (size_t)2 * NN * NOUTP;      // [N*64]
    float* rbuf = ws + (size_t)3 * NN * NOUTP;      // [N*10]
    float* muA  = rbuf + (size_t)NN * KCLUS;        // [640]
    float* muB  = muA + KCLUS * NOUTP;              // [640]
    float* acc  = muB + KCLUS * NOUTP;              // [640]
    float* rsum = acc + KCLUS * NOUTP;              // [16] (contiguous with acc)

    // Packed bf16 weights live ABOVE the agg1 region (survive whole pipeline).
    __bf16* W1p = (__bf16*)(ws + (size_t)2 * NN * NHID);            // 32*4*32*16
    __bf16* W2p = W1p + (size_t)(NHID / 16) * (NFEAT / 32) * 512;   // 4*16*32*16

    // ---- pack weights into WMMA fragment order -----------------------------
    {
        const int tot1 = (NHID / 16) * (NFEAT / 32) * 32;   // 4096 fragments
        const int tot2 = (NOUTP / 16) * (NHID / 32) * 32;   // 2048 fragments
        pack_b_frag<<<(tot1 + 255) / 256, 256, 0, stream>>>(W1, W1p, NFEAT / 32, NHID, NHID, tot1);
        pack_b_frag<<<(tot2 + 255) / 256, 256, 0, stream>>>(W2, W2p, NHID / 32, NOUT, NOUT, tot2);
    }

    // ---- Layer 1: sup1 = x @ W1 (bf16 WMMA, f32 accumulate) ----------------
    gemm_wmma_bf16<NFEAT, 4, 8><<<dim3(NN / 16, 1), 256, 0, stream>>>(x, W1p, sup1, NHID);
    hipMemsetAsync(agg1, 0, (size_t)NN * NHID * sizeof(float), stream);
    spmm_atomic<<<E, 256, 0, stream>>>(erows, ecols, evals, sup1, agg1, NHID, NHID);
    bias_relu_inplace<<<(int)(((long)NN * NHID + 255) / 256), 256, 0, stream>>>(
        agg1, b1, (long)NN * NHID, NHID - 1);

    // ---- Layer 2: sup2 = h @ W2 (N padded 50->64) --------------------------
    gemm_wmma_bf16<NHID, 1, 4><<<dim3(NN / 16, 1), 128, 0, stream>>>(agg1, W2p, sup2, NOUTP);
    hipMemsetAsync(agg2, 0, (size_t)NN * NOUTP * sizeof(float), stream);
    spmm_atomic<<<E, 64, 0, stream>>>(erows, ecols, evals, sup2, agg2, NOUTP, NOUTP);

    // ---- embeds (+bias) -> d_out; row-normalized data ----------------------
    finalize_embeds<<<NN / 8, 256, 0, stream>>>(agg2, b2, out_emb, data, NN);

    // ---- clustering: mu0 -> (step) -> mu1 -> (step) -> mu2 -----------------
    mu_pad<<<1, 640, 0, stream>>>(mu0, muA);

    dist_softmax<<<(NN + 255) / 256, 256, 0, stream>>>(data, muA, rbuf, nullptr, NN);
    hipMemsetAsync(acc, 0, (KCLUS * NOUTP + 16) * sizeof(float), stream);
    mu_reduce<<<256, 640, 0, stream>>>(rbuf, data, acc, rsum, NN);
    mu_finalize<<<1, 640, 0, stream>>>(acc, rsum, muB);

    dist_softmax<<<(NN + 255) / 256, 256, 0, stream>>>(data, muB, rbuf, nullptr, NN);
    hipMemsetAsync(acc, 0, (KCLUS * NOUTP + 16) * sizeof(float), stream);
    mu_reduce<<<256, 640, 0, stream>>>(rbuf, data, acc, rsum, NN);
    mu_finalize<<<1, 640, 0, stream>>>(acc, rsum, muA);   // muA = mu2 (mu_out)

    // ---- final assignments with mu2 ---------------------------------------
    dist_softmax<<<(NN + 255) / 256, 256, 0, stream>>>(data, muA, out_r, out_dst, NN);
    mu_unpad<<<1, 512, 0, stream>>>(muA, out_mu);
}
